// _HMoEModule_68478958567429
// MI455X (gfx1250) — matrix operand order repaired and checked by reading, not compiled, not run on hardware
//
#include <hip/hip_runtime.h>
#include <hip/hip_bf16.h>

// ---------------------------------------------------------------------------
// HMoE forward for MI455X (gfx1250, wave32, WMMA bf16 16x16x32).
// Sizes: B=4096, F=1024, GH=256, EH=512, C=1000(pad 1024), 15 gates, 16 experts
// ---------------------------------------------------------------------------

typedef __attribute__((ext_vector_type(16))) __bf16 v16bf;
typedef __attribute__((ext_vector_type(8)))  float v8f;

#define BSZ   4096
#define FDIM  1024
#define GHD   256
#define EHD   512
#define CDIM  1000
#define CPAD  1024
#define NGATE 15
#define NEXP  16

union Frag { v16bf v; uint4 q[2]; };

__device__ __forceinline__ unsigned short f2bf(float f) {
  unsigned u = __float_as_uint(f);
  unsigned r = 0x7FFFu + ((u >> 16) & 1u);   // round-to-nearest-even
  return (unsigned short)((u + r) >> 16);
}

// Wave-level bf16 GEMM: acc[MT][NT] (16x16 tiles) += A[row0..][K] * B[col0..][K]^T
// A, B stored K-major (lda/ldb = K stride in elements). K multiple of 32.
// Fragment layouts follow CDNA5 ISA 7.12.2 (16-bit A 16x32, B 32x16).
template<int MT, int NT>
__device__ __forceinline__ void wave_gemm(const unsigned short* __restrict__ A, int lda, int row0,
                                          const unsigned short* __restrict__ Bm, int ldb, int col0,
                                          int K, v8f (&acc)[MT][NT]) {
  const int lane = threadIdx.x & 31;
  const int l16  = lane & 15;
  const int half = lane >> 4;
  for (int k0 = 0; k0 < K; k0 += 32) {
    Frag a[MT];
#pragma unroll
    for (int mt = 0; mt < MT; ++mt) {
      const unsigned short* pa = A + (size_t)(row0 + mt * 16 + l16) * lda + k0 + half * 8;
      a[mt].q[0] = *reinterpret_cast<const uint4*>(pa);        // K = k0+half*8 .. +7
      a[mt].q[1] = *reinterpret_cast<const uint4*>(pa + 16);   // K = k0+16+half*8 .. +7
    }
#pragma unroll
    for (int nt = 0; nt < NT; ++nt) {
      Frag b;
      const unsigned short* pb = Bm + (size_t)(col0 + nt * 16 + l16) * ldb + k0 + half * 8;
      b.q[0] = *reinterpret_cast<const uint4*>(pb);            // K = k0+half*8 .. +7
      b.q[1] = *reinterpret_cast<const uint4*>(pb + 16);       // K = k0+16+half*8 .. +7
#pragma unroll
      for (int mt = 0; mt < MT; ++mt)
        acc[mt][nt] = __builtin_amdgcn_wmma_f32_16x16x32_bf16(
            false, a[mt].v, false, b.v, (short)0, acc[mt][nt], false, false);
    }
  }
}

// ---------------------------------------------------------------------------
// conversion kernels
// ---------------------------------------------------------------------------
__global__ __launch_bounds__(256) void k_f32_to_bf16(const float* __restrict__ s,
                                                     unsigned short* __restrict__ d, int n) {
  int i = blockIdx.x * 256 + threadIdx.x;
  if (i < n) d[i] = f2bf(s[i]);
}

// ew2 [16][1000][512] f32 -> [16][1024][512] bf16, zero-padded classes
__global__ __launch_bounds__(256) void k_ew2_pad(const float* __restrict__ s,
                                                 unsigned short* __restrict__ d) {
  int i = blockIdx.x * 256 + threadIdx.x;      // over 16*1024*512
  int k = i & 511;
  int c = (i >> 9) & 1023;
  int e = i >> 19;
  d[i] = (c < CDIM) ? f2bf(s[((size_t)e * CDIM + c) * EHD + k]) : (unsigned short)0;
}

__global__ __launch_bounds__(256) void k_zero(float* __restrict__ p, int n) {
  int i = blockIdx.x * 256 + threadIdx.x;
  if (i < n) p[i] = 0.f;
}

// ---------------------------------------------------------------------------
// gate hidden: gh[g][b][h] = tanh(x . gw1[g]^T + gb1[g])   (f32 out)
// waves = 15 * (4096/32) * (256/64) = 7680 ; 960 blocks * 8 waves
// ---------------------------------------------------------------------------
__global__ __launch_bounds__(256) void k_gate_gemm(const unsigned short* __restrict__ xb,
                                                   const unsigned short* __restrict__ gw1b,
                                                   const float* __restrict__ gb1,
                                                   float* __restrict__ gh) {
  int wid  = blockIdx.x * 8 + (threadIdx.x >> 5);
  int g    = wid >> 9;
  int rem  = wid & 511;
  int row0 = (rem >> 2) * 32;
  int col0 = (rem & 3) * 64;
  const int lane = threadIdx.x & 31, l16 = lane & 15, half = lane >> 4;

  v8f acc[2][4];
  v8f z = {0.f, 0.f, 0.f, 0.f, 0.f, 0.f, 0.f, 0.f};
#pragma unroll
  for (int mt = 0; mt < 2; ++mt)
#pragma unroll
    for (int nt = 0; nt < 4; ++nt) acc[mt][nt] = z;

  wave_gemm<2, 4>(xb, FDIM, row0, gw1b + (size_t)g * GHD * FDIM, FDIM, col0, FDIM, acc);

#pragma unroll
  for (int mt = 0; mt < 2; ++mt)
#pragma unroll
    for (int nt = 0; nt < 4; ++nt) {
      int col = col0 + nt * 16 + l16;
      float bv = gb1[g * GHD + col];
#pragma unroll
      for (int j = 0; j < 8; ++j) {
        int row = row0 + mt * 16 + half * 8 + j;
        gh[((size_t)g * BSZ + row) * GHD + col] = tanhf(acc[mt][nt][j] + bv);
      }
    }
}

// ---------------------------------------------------------------------------
// gate logits + 2-way softmax + tree product -> wleaf[b][16]
// one wave per batch row; 512 blocks * 8 waves
// ---------------------------------------------------------------------------
__global__ __launch_bounds__(256) void k_gate_tree(const float* __restrict__ gh,
                                                   const float* __restrict__ gw2,
                                                   const float* __restrict__ gb2,
                                                   float* __restrict__ wleaf) {
  int wv = threadIdx.x >> 5;
  int b  = blockIdx.x * 8 + wv;
  int lane = threadIdx.x & 31;
  __shared__ float pbuf[8][16];

  for (int g = 0; g < NGATE; ++g) {
    float a0 = 0.f, a1 = 0.f;
    const float* gb = gh + ((size_t)g * BSZ + b) * GHD;
    const float* w0 = gw2 + (g * 2 + 0) * GHD;
    const float* w1 = gw2 + (g * 2 + 1) * GHD;
    for (int h = lane; h < GHD; h += 32) {
      float v = gb[h];
      a0 += v * w0[h];
      a1 += v * w1[h];
    }
#pragma unroll
    for (int off = 16; off > 0; off >>= 1) {
      a0 += __shfl_xor(a0, off, 32);
      a1 += __shfl_xor(a1, off, 32);
    }
    float l0 = a0 + gb2[g * 2], l1 = a1 + gb2[g * 2 + 1];
    float m = fmaxf(l0, l1);
    float e0 = __expf(l0 - m), e1 = __expf(l1 - m);
    if (lane == 0) pbuf[wv][g] = e0 / (e0 + e1);   // P(child 0)
  }
  __syncthreads();
  if (lane < NEXP) {
    float w = 1.f;
    int idx = 0;
#pragma unroll
    for (int d = 0; d < 4; ++d) {
      int c = (lane >> (3 - d)) & 1;
      int g = (1 << d) - 1 + idx;
      float p0 = pbuf[wv][g];
      w *= c ? (1.f - p0) : p0;
      idx = idx * 2 + c;
    }
    wleaf[(size_t)b * NEXP + lane] = w;
  }
}

// ---------------------------------------------------------------------------
// expert hidden: eh[e][b][h] = relu(x . ew1[e]^T + eb1[e])  (bf16 out)
// waves = 16 * (4096/32) * (512/64) = 16384 ; 2048 blocks
// ---------------------------------------------------------------------------
__global__ __launch_bounds__(256) void k_expert_h(const unsigned short* __restrict__ xb,
                                                  const unsigned short* __restrict__ ew1b,
                                                  const float* __restrict__ eb1,
                                                  unsigned short* __restrict__ eh) {
  int wid  = blockIdx.x * 8 + (threadIdx.x >> 5);
  int e    = wid >> 10;
  int rem  = wid & 1023;
  int row0 = (rem >> 3) * 32;
  int col0 = (rem & 7) * 64;
  const int lane = threadIdx.x & 31, l16 = lane & 15, half = lane >> 4;

  v8f acc[2][4];
  v8f z = {0.f, 0.f, 0.f, 0.f, 0.f, 0.f, 0.f, 0.f};
#pragma unroll
  for (int mt = 0; mt < 2; ++mt)
#pragma unroll
    for (int nt = 0; nt < 4; ++nt) acc[mt][nt] = z;

  wave_gemm<2, 4>(xb, FDIM, row0, ew1b + (size_t)e * EHD * FDIM, FDIM, col0, FDIM, acc);

#pragma unroll
  for (int mt = 0; mt < 2; ++mt)
#pragma unroll
    for (int nt = 0; nt < 4; ++nt) {
      int col = col0 + nt * 16 + l16;
      float bv = eb1[e * EHD + col];
#pragma unroll
      for (int j = 0; j < 8; ++j) {
        int row = row0 + mt * 16 + half * 8 + j;
        float v = fmaxf(acc[mt][nt][j] + bv, 0.f);
        eh[((size_t)e * BSZ + row) * EHD + col] = f2bf(v);
      }
    }
}

// ---------------------------------------------------------------------------
// expert out + softmax + mixture:
//   out[b][c] += sum_e wleaf[b][e] * softmax_c(eh[e][b] . ew2[e]^T + eb2[e])
// block = 16 rows x 4 experts (looped), 8 waves each owning 128 columns.
// grid = 4 expert-groups * 256 row-tiles = 1024 blocks.
// ---------------------------------------------------------------------------
__global__ __launch_bounds__(256) void k_expert_out_mix(const unsigned short* __restrict__ eh,
                                                        const unsigned short* __restrict__ ew2b,
                                                        const float* __restrict__ eb2,
                                                        const float* __restrict__ wleaf,
                                                        float* __restrict__ out) {
  const int tid  = threadIdx.x;
  const int wave = tid >> 5, lane = tid & 31, l16 = lane & 15, half = lane >> 4;
  const int eg   = blockIdx.x >> 8;        // expert group 0..3
  const int b0   = (blockIdx.x & 255) * 16;
  const int n0   = wave * 128;

  __shared__ float maxbuf[16][8];
  __shared__ float sumbuf[16][8];

  float mix[8][8];
#pragma unroll
  for (int nt = 0; nt < 8; ++nt)
#pragma unroll
    for (int j = 0; j < 8; ++j) mix[nt][j] = 0.f;

  for (int ei = 0; ei < 4; ++ei) {
    int e = eg * 4 + ei;

    v8f acc[1][8];
    v8f z = {0.f, 0.f, 0.f, 0.f, 0.f, 0.f, 0.f, 0.f};
#pragma unroll
    for (int nt = 0; nt < 8; ++nt) acc[0][nt] = z;

    wave_gemm<1, 8>(eh + (size_t)e * BSZ * EHD, EHD, b0,
                    ew2b + (size_t)e * CPAD * EHD, EHD, n0, EHD, acc);

    // bias + mask padded classes
#pragma unroll
    for (int nt = 0; nt < 8; ++nt) {
      int n = n0 + nt * 16 + l16;
      float bv = (n < CDIM) ? eb2[e * CDIM + n] : 0.f;
#pragma unroll
      for (int j = 0; j < 8; ++j)
        acc[0][nt][j] = (n < CDIM) ? (acc[0][nt][j] + bv) : -3.0e38f;
    }

    // row max: in-lane over nt, 16-lane shuffle, cross-wave via LDS
    float rmax[8];
#pragma unroll
    for (int j = 0; j < 8; ++j) {
      float m = acc[0][0][j];
#pragma unroll
      for (int nt = 1; nt < 8; ++nt) m = fmaxf(m, acc[0][nt][j]);
      rmax[j] = m;
    }
#pragma unroll
    for (int off = 1; off < 16; off <<= 1)
#pragma unroll
      for (int j = 0; j < 8; ++j) rmax[j] = fmaxf(rmax[j], __shfl_xor(rmax[j], off, 32));
    if (l16 == 0)
#pragma unroll
      for (int j = 0; j < 8; ++j) maxbuf[half * 8 + j][wave] = rmax[j];
    __syncthreads();
#pragma unroll
    for (int j = 0; j < 8; ++j) {
      float m = maxbuf[half * 8 + j][0];
#pragma unroll
      for (int w = 1; w < 8; ++w) m = fmaxf(m, maxbuf[half * 8 + j][w]);
      rmax[j] = m;
    }

    // exp + row sum
    float rsum[8];
#pragma unroll
    for (int j = 0; j < 8; ++j) rsum[j] = 0.f;
#pragma unroll
    for (int nt = 0; nt < 8; ++nt)
#pragma unroll
      for (int j = 0; j < 8; ++j) {
        float ex = __expf(acc[0][nt][j] - rmax[j]);   // masked -> 0
        acc[0][nt][j] = ex;
        rsum[j] += ex;
      }
#pragma unroll
    for (int off = 1; off < 16; off <<= 1)
#pragma unroll
      for (int j = 0; j < 8; ++j) rsum[j] += __shfl_xor(rsum[j], off, 32);
    if (l16 == 0)
#pragma unroll
      for (int j = 0; j < 8; ++j) sumbuf[half * 8 + j][wave] = rsum[j];
    __syncthreads();
#pragma unroll
    for (int j = 0; j < 8; ++j) {
      float s = 0.f;
#pragma unroll
      for (int w = 0; w < 8; ++w) s += sumbuf[half * 8 + j][w];
      rsum[j] = s;
    }

    // accumulate leaf-weighted probabilities
    float wv[8];
#pragma unroll
    for (int j = 0; j < 8; ++j)
      wv[j] = wleaf[(size_t)(b0 + half * 8 + j) * NEXP + e] / rsum[j];
#pragma unroll
    for (int nt = 0; nt < 8; ++nt)
#pragma unroll
      for (int j = 0; j < 8; ++j) mix[nt][j] += wv[j] * acc[0][nt][j];
  }

  // one atomic add per element per expert-group (out pre-zeroed)
#pragma unroll
  for (int nt = 0; nt < 8; ++nt) {
    int n = n0 + nt * 16 + l16;
    if (n < CDIM) {
#pragma unroll
      for (int j = 0; j < 8; ++j)
        atomicAdd(&out[(size_t)(b0 + half * 8 + j) * CDIM + n], mix[nt][j]);
    }
  }
}

// ---------------------------------------------------------------------------
// launch
// ---------------------------------------------------------------------------
extern "C" void kernel_launch(void* const* d_in, const int* in_sizes, int n_in,
                              void* d_out, int out_size, void* d_ws, size_t ws_size,
                              hipStream_t stream) {
  const float* x   = (const float*)d_in[0];
  const float* gw1 = (const float*)d_in[1];
  const float* gb1 = (const float*)d_in[2];
  const float* gw2 = (const float*)d_in[3];
  const float* gb2 = (const float*)d_in[4];
  const float* ew1 = (const float*)d_in[5];
  const float* eb1 = (const float*)d_in[6];
  const float* ew2 = (const float*)d_in[7];
  const float* eb2 = (const float*)d_in[8];
  float* out = (float*)d_out;

  // workspace carve-up (~180 MB)
  unsigned short* xb   = (unsigned short*)d_ws;                 // 4096*1024
  unsigned short* gw1b = xb   + (size_t)BSZ * FDIM;             // 15*256*1024
  unsigned short* ew1b = gw1b + (size_t)NGATE * GHD * FDIM;     // 16*512*1024
  unsigned short* ew2b = ew1b + (size_t)NEXP * EHD * FDIM;      // 16*1024*512
  float* gh   = (float*)(ew2b + (size_t)NEXP * CPAD * EHD);     // 15*4096*256 f32
  float* wlf  = gh + (size_t)NGATE * BSZ * GHD;                 // 4096*16 f32
  unsigned short* eh = (unsigned short*)(wlf + (size_t)BSZ * NEXP); // 16*4096*512

  const int nx   = BSZ * FDIM;
  const int ngw1 = NGATE * GHD * FDIM;
  const int new1 = NEXP * EHD * FDIM;
  const int new2p = NEXP * CPAD * EHD;

  k_f32_to_bf16<<<(nx   + 255) / 256, 256, 0, stream>>>(x,   xb,   nx);
  k_f32_to_bf16<<<(ngw1 + 255) / 256, 256, 0, stream>>>(gw1, gw1b, ngw1);
  k_f32_to_bf16<<<(new1 + 255) / 256, 256, 0, stream>>>(ew1, ew1b, new1);
  k_ew2_pad<<<new2p / 256, 256, 0, stream>>>(ew2, ew2b);

  k_gate_gemm<<<NGATE * (BSZ / 32) * (GHD / 64) / 8, 256, 0, stream>>>(xb, gw1b, gb1, gh);
  k_gate_tree<<<BSZ / 8, 256, 0, stream>>>(gh, gw2, gb2, wlf);
  k_expert_h<<<NEXP * (BSZ / 32) * (EHD / 64) / 8, 256, 0, stream>>>(xb, ew1b, eb1, eh);

  k_zero<<<(BSZ * CDIM + 255) / 256, 256, 0, stream>>>(out, BSZ * CDIM);
  k_expert_out_mix<<<4 * (BSZ / 16), 256, 0, stream>>>(eh, ew2b, eb2, wlf, out);
}
